// GloAN_44504451121485
// MI455X (gfx1250) — matrix operand order repaired and checked by reading, not compiled, not run on hardware
//
#include <hip/hip_runtime.h>
#include <hip/hip_bf16.h>
#include <math.h>

typedef __attribute__((ext_vector_type(16))) _Float16 v16h;
typedef __attribute__((ext_vector_type(8)))  float    v8f;

union Frag16 { v16h v; float4 f4[2]; };

#define B_SZ  32
#define C_IN  128
#define NTOK  3136
#define DDIM  64
#define EPSLN 1e-5f

// ---------------------------------------------------------------------------
// K1: embed  t = LN2(LN1(x^T @ We + be))   -> Tg [B, N, 64] f32
// ---------------------------------------------------------------------------
__global__ __launch_bounds__(256) void k_embed(
    const float* __restrict__ x,  const float* __restrict__ We, const float* __restrict__ be,
    const float* __restrict__ g1, const float* __restrict__ b1,
    const float* __restrict__ g2, const float* __restrict__ b2,
    float* __restrict__ Tg)
{
    __shared__ float sWe[C_IN * DDIM];                    // 32 KB
    __shared__ float sbe[DDIM], sg1[DDIM], sb1[DDIM], sg2[DDIM], sb2[DDIM];
    const int tid = threadIdx.x;
    for (int i = tid; i < C_IN * DDIM; i += 256) sWe[i] = We[i];
    if (tid < DDIM) {
        sbe[tid] = be[tid]; sg1[tid] = g1[tid]; sb1[tid] = b1[tid];
        sg2[tid] = g2[tid]; sb2[tid] = b2[tid];
    }
    __syncthreads();

    const int ntiles = (NTOK + 255) / 256;                // 13
    const int b    = blockIdx.x / ntiles;
    const int tile = blockIdx.x % ntiles;
    const int n    = tile * 256 + tid;
    if (n >= NTOK) return;

    const float* xb = x + (size_t)b * C_IN * NTOK + n;

    float t[DDIM];
    #pragma unroll
    for (int d = 0; d < DDIM; ++d) t[d] = sbe[d];
    for (int c = 0; c < C_IN; ++c) {
        float xv = xb[(size_t)c * NTOK];
        #pragma unroll
        for (int d = 0; d < DDIM; ++d) t[d] = fmaf(xv, sWe[c * DDIM + d], t[d]);
    }
    // two layer norms
    #pragma unroll 1
    for (int pass = 0; pass < 2; ++pass) {
        const float* gg = pass ? sg2 : sg1;
        const float* bb = pass ? sb2 : sb1;
        float mu = 0.f;
        #pragma unroll
        for (int d = 0; d < DDIM; ++d) mu += t[d];
        mu *= (1.0f / DDIM);
        float var = 0.f;
        #pragma unroll
        for (int d = 0; d < DDIM; ++d) { float z = t[d] - mu; var = fmaf(z, z, var); }
        var *= (1.0f / DDIM);
        float r = rsqrtf(var + EPSLN);
        #pragma unroll
        for (int d = 0; d < DDIM; ++d) t[d] = (t[d] - mu) * r * gg[d] + bb[d];
    }
    float* trow = Tg + ((size_t)b * NTOK + n) * DDIM;
    #pragma unroll
    for (int d = 0; d < DDIM; ++d) trow[d] = t[d];
}

// ---------------------------------------------------------------------------
// K2: qkv = t @ Wqkv + bqkv ; Q (pre-scaled by 0.125) & K f16 token-major,
//     V stored transposed (feature-major) for WMMA B-fragment loads.
// ---------------------------------------------------------------------------
__global__ __launch_bounds__(256) void k_qkv(
    const float* __restrict__ Tg, const float* __restrict__ Wqkv, const float* __restrict__ bqkv,
    _Float16* __restrict__ Qg, _Float16* __restrict__ Kg, _Float16* __restrict__ Vtg)
{
    __shared__ float sW[DDIM * 192];                      // 48 KB
    __shared__ float sb[192];
    const int tid = threadIdx.x;
    for (int i = tid; i < DDIM * 192; i += 256) sW[i] = Wqkv[i];
    if (tid < 192) sb[tid] = bqkv[tid];
    __syncthreads();

    const int idx = blockIdx.x * 256 + tid;               // b*N + n
    if (idx >= B_SZ * NTOK) return;
    const int b = idx / NTOK;
    const int n = idx - b * NTOK;

    float t[DDIM];
    const float* trow = Tg + (size_t)idx * DDIM;
    #pragma unroll
    for (int d = 0; d < DDIM; ++d) t[d] = trow[d];

    const size_t toff = (size_t)idx * DDIM;
    #pragma unroll 1
    for (int j = 0; j < DDIM; ++j) {
        float s = sb[j];
        #pragma unroll
        for (int d = 0; d < DDIM; ++d) s = fmaf(t[d], sW[d * 192 + j], s);
        Qg[toff + j] = (_Float16)(s * 0.125f);
    }
    #pragma unroll 1
    for (int j = 0; j < DDIM; ++j) {
        float s = sb[64 + j];
        #pragma unroll
        for (int d = 0; d < DDIM; ++d) s = fmaf(t[d], sW[d * 192 + 64 + j], s);
        Kg[toff + j] = (_Float16)s;
    }
    #pragma unroll 1
    for (int j = 0; j < DDIM; ++j) {
        float s = sb[128 + j];
        #pragma unroll
        for (int d = 0; d < DDIM; ++d) s = fmaf(t[d], sW[d * 192 + 128 + j], s);
        Vtg[((size_t)b * DDIM + j) * NTOK + n] = (_Float16)s;
    }
}

// ---------------------------------------------------------------------------
// K3: flash attention. 4 waves/block, 1 query tile (16 rows) per wave,
//     64 queries per block amortize each 32-key block (K/V frags hit WGP$).
//     v_wmma_f32_16x16x32_f16 throughout; ~170 live VGPRs/wave (no spills).
// ---------------------------------------------------------------------------
__global__ __launch_bounds__(128) void k_attn(
    const _Float16* __restrict__ Qg, const _Float16* __restrict__ Kg,
    const _Float16* __restrict__ Vtg, float* __restrict__ Og)
{
    __shared__ __align__(16) _Float16 sP[4][16 * 40];     // per-wave P bounce (5 KB)

    const int lane = threadIdx.x & 31;
    const int wave = threadIdx.x >> 5;
    const int sel  = lane >> 4;        // half-wave selector
    const int lid  = lane & 15;
    const int nqb  = NTOK / 64;        // 49 query blocks of 64
    const int b    = blockIdx.x / nqb;
    const int q0   = (blockIdx.x % nqb) * 64 + wave * 16; // this wave's query tile

    // Q A-fragments (feature halves ko=0,32), kept in registers.
    Frag16 aq[2];
    {
        const _Float16* qrow = Qg + ((size_t)b * NTOK + q0 + lid) * DDIM;
        #pragma unroll
        for (int koi = 0; koi < 2; ++koi) {
            const int ko = koi * 32;
            aq[koi].f4[0] = *(const float4*)(qrow + ko + sel * 16);
            aq[koi].f4[1] = *(const float4*)(qrow + ko + sel * 16 + 8);
        }
    }

    v8f   Oacc[4];
    float mrun[8], lrun[8];
    #pragma unroll
    for (int nt = 0; nt < 4; ++nt)
        #pragma unroll
        for (int j = 0; j < 8; ++j) Oacc[nt][j] = 0.0f;
    #pragma unroll
    for (int j = 0; j < 8; ++j) { mrun[j] = -1e30f; lrun[j] = 0.0f; }

    v8f z8;
    #pragma unroll
    for (int j = 0; j < 8; ++j) z8[j] = 0.0f;

    _Float16* myP = &sP[wave][0];

    #pragma unroll 1
    for (int kb = 0; kb < NTOK; kb += 32) {
        // K B-fragments: bk[kt][koi] covers keys kb+16*kt.., features koi*32..
        Frag16 bk[2][2];
        #pragma unroll
        for (int kt = 0; kt < 2; ++kt) {
            const _Float16* krow = Kg + ((size_t)b * NTOK + kb + kt * 16 + lid) * DDIM;
            #pragma unroll
            for (int koi = 0; koi < 2; ++koi) {
                const int off = koi * 32 + sel * 16;
                bk[kt][koi].f4[0] = *(const float4*)(krow + off);
                bk[kt][koi].f4[1] = *(const float4*)(krow + off + 8);
            }
        }
        // V^T B-fragments: bv[nt] covers out features nt*16.., keys kb..kb+31
        Frag16 bv[4];
        #pragma unroll
        for (int nt = 0; nt < 4; ++nt) {
            const _Float16* vrow = Vtg + ((size_t)b * DDIM + nt * 16 + lid) * NTOK + kb + sel * 16;
            bv[nt].f4[0] = *(const float4*)(vrow);
            bv[nt].f4[1] = *(const float4*)(vrow + 8);
        }

        // S = Q @ K^T  (16x32 scores: two 16x16 C fragments)
        v8f S[2];
        #pragma unroll
        for (int kt = 0; kt < 2; ++kt) {
            S[kt] = __builtin_amdgcn_wmma_f32_16x16x32_f16(
                        false, aq[0].v, false, bk[kt][0].v, (short)0, z8, false, false);
            S[kt] = __builtin_amdgcn_wmma_f32_16x16x32_f16(
                        false, aq[1].v, false, bk[kt][1].v, (short)0, S[kt], false, false);
        }

        // online softmax (rows live in 16-lane halves; xor masks 1,2,4,8 stay in-half)
        float pscale[8];
        #pragma unroll
        for (int j = 0; j < 8; ++j) {
            float rm = fmaxf(S[0][j], S[1][j]);
            rm = fmaxf(rm, __shfl_xor(rm, 1, 32));
            rm = fmaxf(rm, __shfl_xor(rm, 2, 32));
            rm = fmaxf(rm, __shfl_xor(rm, 4, 32));
            rm = fmaxf(rm, __shfl_xor(rm, 8, 32));
            float mn = fmaxf(mrun[j], rm);
            float sc = __expf(mrun[j] - mn);
            mrun[j] = mn;
            pscale[j] = sc;
            float p0 = __expf(S[0][j] - mn);
            float p1 = __expf(S[1][j] - mn);
            S[0][j] = p0; S[1][j] = p1;
            float rs = p0 + p1;
            rs += __shfl_xor(rs, 1, 32);
            rs += __shfl_xor(rs, 2, 32);
            rs += __shfl_xor(rs, 4, 32);
            rs += __shfl_xor(rs, 8, 32);
            lrun[j] = lrun[j] * sc + rs;
        }
        #pragma unroll
        for (int nt = 0; nt < 4; ++nt)
            #pragma unroll
            for (int j = 0; j < 8; ++j) Oacc[nt][j] *= pscale[j];

        // P: C-layout -> f16 A-layout via per-wave LDS bounce
        #pragma unroll
        for (int j = 0; j < 8; ++j) {
            const int row = j + sel * 8;
            myP[row * 40 + lid]      = (_Float16)S[0][j];
            myP[row * 40 + 16 + lid] = (_Float16)S[1][j];
        }
        __syncthreads();
        Frag16 pa;
        {
            const _Float16* pr = myP + lid * 40;
            pa.f4[0] = *(const float4*)(pr + sel * 16);
            pa.f4[1] = *(const float4*)(pr + sel * 16 + 8);
        }
        __syncthreads();

        // O += P @ V
        #pragma unroll
        for (int nt = 0; nt < 4; ++nt)
            Oacc[nt] = __builtin_amdgcn_wmma_f32_16x16x32_f16(
                false, pa.v, false, bv[nt].v, (short)0, Oacc[nt], false, false);
    }

    // normalize and write O [B, N, 64] f32
    float inv[8];
    #pragma unroll
    for (int j = 0; j < 8; ++j) inv[j] = 1.0f / lrun[j];
    #pragma unroll
    for (int nt = 0; nt < 4; ++nt)
        #pragma unroll
        for (int j = 0; j < 8; ++j) {
            const int row = q0 + sel * 8 + j;
            Og[((size_t)b * NTOK + row) * DDIM + nt * 16 + lid] = Oacc[nt][j] * inv[j];
        }
}

// ---------------------------------------------------------------------------
// K4: proj + LN + MLP(gelu) + mean over D  -> out [B, 1, H, W]
// ---------------------------------------------------------------------------
__global__ __launch_bounds__(256) void k_mlp(
    const float* __restrict__ Og, const float* __restrict__ Wp, const float* __restrict__ bp,
    const float* __restrict__ gm, const float* __restrict__ bm,
    const float* __restrict__ W1, const float* __restrict__ bm1,
    const float* __restrict__ W2, const float* __restrict__ bm2,
    float* __restrict__ out)
{
    __shared__ float sWp[DDIM * DDIM], sW1[DDIM * DDIM], sW2[DDIM * DDIM];   // 48 KB
    __shared__ float sbp[DDIM], sgm[DDIM], sbm[DDIM], sb1[DDIM], sb2[DDIM];
    const int tid = threadIdx.x;
    for (int i = tid; i < DDIM * DDIM; i += 256) {
        sWp[i] = Wp[i]; sW1[i] = W1[i]; sW2[i] = W2[i];
    }
    if (tid < DDIM) {
        sbp[tid] = bp[tid]; sgm[tid] = gm[tid]; sbm[tid] = bm[tid];
        sb1[tid] = bm1[tid]; sb2[tid] = bm2[tid];
    }
    __syncthreads();

    const int idx = blockIdx.x * 256 + tid;               // b*N + n
    if (idx >= B_SZ * NTOK) return;

    float o[DDIM];
    const float* orow = Og + (size_t)idx * DDIM;
    #pragma unroll
    for (int d = 0; d < DDIM; ++d) o[d] = orow[d];

    // proj
    float t[DDIM];
    #pragma unroll 1
    for (int j = 0; j < DDIM; ++j) {
        float s = sbp[j];
        #pragma unroll
        for (int d = 0; d < DDIM; ++d) s = fmaf(o[d], sWp[d * DDIM + j], s);
        t[j] = s;
    }
    // LN
    {
        float mu = 0.f;
        #pragma unroll
        for (int d = 0; d < DDIM; ++d) mu += t[d];
        mu *= (1.0f / DDIM);
        float var = 0.f;
        #pragma unroll
        for (int d = 0; d < DDIM; ++d) { float z = t[d] - mu; var = fmaf(z, z, var); }
        var *= (1.0f / DDIM);
        float r = rsqrtf(var + EPSLN);
        #pragma unroll
        for (int d = 0; d < DDIM; ++d) t[d] = (t[d] - mu) * r * sgm[d] + sbm[d];
    }
    // h = gelu(t @ W1 + b1)
    #pragma unroll 1
    for (int j = 0; j < DDIM; ++j) {
        float s = sb1[j];
        #pragma unroll
        for (int d = 0; d < DDIM; ++d) s = fmaf(t[d], sW1[d * DDIM + j], s);
        o[j] = 0.5f * s * (1.0f + erff(s * 0.70710678118654752f));
    }
    // out token = mean_j (h @ W2 + b2)[j]
    float acc = 0.f;
    #pragma unroll 1
    for (int j = 0; j < DDIM; ++j) {
        float s = sb2[j];
        #pragma unroll
        for (int d = 0; d < DDIM; ++d) s = fmaf(o[d], sW2[d * DDIM + j], s);
        acc += s;
    }
    out[idx] = acc * (1.0f / DDIM);
}

// ---------------------------------------------------------------------------
extern "C" void kernel_launch(void* const* d_in, const int* in_sizes, int n_in,
                              void* d_out, int out_size, void* d_ws, size_t ws_size,
                              hipStream_t stream) {
    const float* x    = (const float*)d_in[0];
    const float* We   = (const float*)d_in[1];
    const float* be   = (const float*)d_in[2];
    const float* g1   = (const float*)d_in[3];
    const float* b1   = (const float*)d_in[4];
    const float* g2   = (const float*)d_in[5];
    const float* b2   = (const float*)d_in[6];
    const float* Wqkv = (const float*)d_in[7];
    const float* bqkv = (const float*)d_in[8];
    const float* Wp   = (const float*)d_in[9];
    const float* bp   = (const float*)d_in[10];
    const float* gm   = (const float*)d_in[11];
    const float* bm   = (const float*)d_in[12];
    const float* W1   = (const float*)d_in[13];
    const float* bm1  = (const float*)d_in[14];
    const float* W2   = (const float*)d_in[15];
    const float* bm2  = (const float*)d_in[16];

    char* ws = (char*)d_ws;
    const size_t QKV_BYTES = (size_t)B_SZ * NTOK * DDIM * sizeof(_Float16); // 12.85 MB each
    _Float16* Qg  = (_Float16*)(ws);
    _Float16* Kg  = (_Float16*)(ws + QKV_BYTES);
    _Float16* Vtg = (_Float16*)(ws + 2 * QKV_BYTES);
    float*    Og  = (float*)   (ws + 3 * QKV_BYTES);   // 25.7 MB; T aliases O (T dead before O written)
    float*    Tg  = Og;

    const int ntok_blocks = (B_SZ * NTOK + 255) / 256;  // 392
    k_embed<<<B_SZ * ((NTOK + 255) / 256), 256, 0, stream>>>(x, We, be, g1, b1, g2, b2, Tg);
    k_qkv  <<<ntok_blocks, 256, 0, stream>>>(Tg, Wqkv, bqkv, Qg, Kg, Vtg);
    k_attn <<<B_SZ * (NTOK / 64), 128, 0, stream>>>(Qg, Kg, Vtg, Og);
    k_mlp  <<<ntok_blocks, 256, 0, stream>>>(Og, Wp, bp, gm, bm, W1, bm1, W2, bm2, (float*)d_out);
}
